// MixtureOfExpertsLayer_29944511988120
// MI455X (gfx1250) — matrix operand order repaired and checked by reading, not compiled, not run on hardware
//
#include <hip/hip_runtime.h>
#include <hip/hip_bf16.h>

// ---------------------------------------------------------------------------
// MoE layer for MI455X (gfx1250): dense 4-expert compute in bf16 WMMA,
// router top-2 + softmax combine. Wave32 everywhere.
// GEMM: A[M,K] @ Bt[N,K]^T with both operands staged in k-contiguous LDS rows
// so every WMMA fragment is two ds_load_b128 ops. Staging uses async
// GLOBAL_LOAD_ASYNC_TO_LDS_B128 when the toolchain exposes the builtins.
// ---------------------------------------------------------------------------

typedef __attribute__((ext_vector_type(16))) __bf16 v16bf;
typedef __attribute__((ext_vector_type(8)))  float  v8f;
typedef int v4i __attribute__((vector_size(16)));           // 128-bit chunk
typedef __attribute__((address_space(1))) v4i* gptr128;     // global
typedef __attribute__((address_space(3))) v4i* lptr128;     // LDS

#define NTOK   8192      // B*S = 4*2048
#define HDIM   1024
#define IDIM   4096

// Async global->LDS staging if the toolchain exposes the gfx1250 builtins.
#if defined(__has_builtin)
#if __has_builtin(__builtin_amdgcn_global_load_async_to_lds_b128) && \
    __has_builtin(__builtin_amdgcn_s_wait_asynccnt)
#define USE_ASYNC_LDS 1
#endif
#endif

// ---------------- bf16 helpers (round-to-nearest-even) ----------------------
__device__ __forceinline__ unsigned short f2bf(float f) {
    unsigned u = __float_as_uint(f);
    u += 0x7FFFu + ((u >> 16) & 1u);
    return (unsigned short)(u >> 16);
}
__device__ __forceinline__ float bf2f(unsigned short h) {
    return __uint_as_float(((unsigned)h) << 16);
}

// ---------------- fp32 -> bf16 conversion (linear) ---------------------------
__global__ __launch_bounds__(256) void convert_bf16_kernel(
    const float* __restrict__ src, unsigned short* __restrict__ dst, size_t n)
{
    size_t i = (size_t)blockIdx.x * blockDim.x + threadIdx.x;
    size_t stride = (size_t)gridDim.x * blockDim.x;
    for (; i < n; i += stride) dst[i] = f2bf(src[i]);
}

// ---------------- fp32 [K,N] -> bf16 [N,K] transposed conversion -------------
// 32x32 tiles through LDS; block = 32x8 threads.
__global__ __launch_bounds__(256) void convert_transpose_kernel(
    const float* __restrict__ src, unsigned short* __restrict__ dst, int N, int K)
{
    __shared__ float tile[32][33];
    const int kb = blockIdx.x * 32, nb = blockIdx.y * 32;
    const int tx = threadIdx.x & 31, ty = threadIdx.x >> 5;   // 32 x 8
#pragma unroll
    for (int r = 0; r < 32; r += 8)
        tile[ty + r][tx] = src[(size_t)(kb + ty + r) * N + nb + tx];
    __syncthreads();
#pragma unroll
    for (int r = 0; r < 32; r += 8)
        dst[(size_t)(nb + ty + r) * K + kb + tx] = f2bf(tile[tx][ty + r]);
}

// ---------------- router: logits -> top2 -> softmax --------------------------
__global__ __launch_bounds__(256) void router_kernel(
    const float* __restrict__ x, const float* __restrict__ rw,
    const float* __restrict__ rb, const float* __restrict__ lb,
    float* __restrict__ outw, int* __restrict__ outsel)
{
    const int token = blockIdx.x * 8 + (threadIdx.x >> 5);
    const int lane  = threadIdx.x & 31;
    const float* xr = x + (size_t)token * HDIM;
    float l0 = 0.f, l1 = 0.f, l2 = 0.f, l3 = 0.f;
    for (int h = lane; h < HDIM; h += 32) {
        float xv = xr[h];
        const float* w = rw + (size_t)h * 4;
        l0 += xv * w[0]; l1 += xv * w[1]; l2 += xv * w[2]; l3 += xv * w[3];
    }
    for (int off = 16; off > 0; off >>= 1) {
        l0 += __shfl_xor(l0, off, 32);
        l1 += __shfl_xor(l1, off, 32);
        l2 += __shfl_xor(l2, off, 32);
        l3 += __shfl_xor(l3, off, 32);
    }
    if (lane == 0) {
        float lg[4] = { l0 + rb[0] + lb[0], l1 + rb[1] + lb[1],
                        l2 + rb[2] + lb[2], l3 + rb[3] + lb[3] };
        int i0 = 0; float m0 = lg[0];
        for (int e = 1; e < 4; ++e) if (lg[e] > m0) { m0 = lg[e]; i0 = e; }
        int i1 = -1; float m1 = -3.4e38f;
        for (int e = 0; e < 4; ++e) if (e != i0 && lg[e] > m1) { m1 = lg[e]; i1 = e; }
        float b = __expf(m1 - m0);
        float s = 1.0f + b;
        outw[token * 2 + 0] = 1.0f / s;
        outw[token * 2 + 1] = b / s;
        outsel[token * 2 + 0] = i0;
        outsel[token * 2 + 1] = i1;
    }
}

// ---------------- bf16 tiled GEMM with fused epilogue ------------------------
// C[M,N] = epi(A[M,K] @ Bt[N,K]^T (+bias)); bf16 in, fp32 accum, bf16 out.
// Tile: BM=128, BN=128, BK=32; 256 threads = 8 waves (4 along M x 2 along N);
// each wave: 2x4 WMMA 16x16 tiles. LDS double-buffered.
#define EPI_NONE      0
#define EPI_BIAS      1
#define EPI_SILU      2
#define EPI_GELU_BIAS 3
#define EPI_RELU_BIAS 4

#define LDS_STRIDE 40    // ushorts per row (32 data + 8 pad); 80B, 16B-aligned

union Frag16 { v16bf v; uint4 q[2]; };

template <int EPI>
__global__ __launch_bounds__(256) void gemm_bf16_kernel(
    const unsigned short* __restrict__ A,    // [M,K]
    const unsigned short* __restrict__ Bt,   // [N,K]  (pre-transposed weights)
    const float* __restrict__ bias,          // [N] or nullptr
    unsigned short* __restrict__ C,          // [M,N]
    int M, int N, int K)
{
    __shared__ __align__(16) unsigned short sA[2][128 * LDS_STRIDE];
    __shared__ __align__(16) unsigned short sB[2][128 * LDS_STRIDE];

    const int tid  = threadIdx.x;
    const int lane = tid & 31;
    const int wave = tid >> 5;
    const int wm   = wave & 3;       // wave row   (4 x 32 rows)
    const int wn   = wave >> 2;      // wave col   (2 x 64 cols)
    const int m0   = blockIdx.y * 128;
    const int n0   = blockIdx.x * 128;
    const int half = lane >> 4;      // 0/1
    const int l16  = lane & 15;

    // staging coordinates for this thread (2 chunks of 16B each for A and B)
    const int row_a[2] = { (tid + 0)   >> 2, (tid + 256) >> 2 };
    const int seg_a[2] = { (tid + 0)   & 3,  (tid + 256) & 3  };

    v8f acc[2][4];
#pragma unroll
    for (int i = 0; i < 2; ++i)
#pragma unroll
        for (int j = 0; j < 4; ++j) { v8f z = {}; acc[i][j] = z; }

#if defined(USE_ASYNC_LDS)
    // ---- async global -> LDS staging (GLOBAL_LOAD_ASYNC_TO_LDS_B128) ------
    auto stage_async = [&](int k0, int buf) {
#pragma unroll
        for (int it = 0; it < 2; ++it) {
            const int row = row_a[it], seg = seg_a[it];
            __builtin_amdgcn_global_load_async_to_lds_b128(
                (gptr128)(A + (size_t)(m0 + row) * K + k0 + seg * 8),
                (lptr128)(&sA[buf][row * LDS_STRIDE + seg * 8]), 0, 0);
            __builtin_amdgcn_global_load_async_to_lds_b128(
                (gptr128)(Bt + (size_t)(n0 + row) * K + k0 + seg * 8),
                (lptr128)(&sB[buf][row * LDS_STRIDE + seg * 8]), 0, 0);
        }
    };
#else
    uint4 ra[2], rb[2];
    auto gload = [&](int k0) {
#pragma unroll
        for (int it = 0; it < 2; ++it) {
            const int row = row_a[it], seg = seg_a[it];
            ra[it] = *(const uint4*)(A  + (size_t)(m0 + row) * K + k0 + seg * 8);
            rb[it] = *(const uint4*)(Bt + (size_t)(n0 + row) * K + k0 + seg * 8);
        }
    };
    auto lstore = [&](int buf) {
#pragma unroll
        for (int it = 0; it < 2; ++it) {
            const int row = row_a[it], seg = seg_a[it];
            *(uint4*)(&sA[buf][row * LDS_STRIDE + seg * 8]) = ra[it];
            *(uint4*)(&sB[buf][row * LDS_STRIDE + seg * 8]) = rb[it];
        }
    };
#endif

    auto compute = [&](int buf) {
        // B fragments: column n holds contiguous k-run [half*16, half*16+16)
        Frag16 bfr[4];
#pragma unroll
        for (int nt = 0; nt < 4; ++nt) {
            const int n = wn * 64 + nt * 16 + l16;
            const unsigned short* p = &sB[buf][n * LDS_STRIDE + half * 16];
            bfr[nt].q[0] = *(const uint4*)(p);
            bfr[nt].q[1] = *(const uint4*)(p + 8);
        }
#pragma unroll
        for (int mt = 0; mt < 2; ++mt) {
            const int m = wm * 32 + mt * 16 + l16;
            // A fragment: contiguous k-run [half*16, half*16+16)
            Frag16 af;
            const unsigned short* p = &sA[buf][m * LDS_STRIDE];
            af.q[0] = *(const uint4*)(p + half * 16);
            af.q[1] = *(const uint4*)(p + half * 16 + 8);
#pragma unroll
            for (int nt = 0; nt < 4; ++nt) {
                acc[mt][nt] = __builtin_amdgcn_wmma_f32_16x16x32_bf16(
                    false, af.v, false, bfr[nt].v,
                    (short)0, acc[mt][nt], false, false);
            }
        }
    };

    // ---- main loop, double buffered -------------------------------------
    int cur = 0;
#if defined(USE_ASYNC_LDS)
    stage_async(0, 0);
    __builtin_amdgcn_s_wait_asynccnt(0);
    __syncthreads();
    for (int k0 = 0; k0 < K; k0 += 32) {
        const bool has_next = (k0 + 32) < K;
        if (has_next) stage_async(k0 + 32, cur ^ 1);   // async into other buffer
        compute(cur);                                  // overlap with compute
        if (has_next) __builtin_amdgcn_s_wait_asynccnt(0);
        __syncthreads();
        cur ^= 1;
    }
#else
    gload(0);
    lstore(0);
    __syncthreads();
    for (int k0 = 0; k0 < K; k0 += 32) {
        const bool has_next = (k0 + 32) < K;
        if (has_next) gload(k0 + 32);         // loads for t+1 in flight
        compute(cur);                         // compute on tile t
        if (has_next) lstore(cur ^ 1);
        __syncthreads();
        cur ^= 1;
    }
#endif

    // ---- epilogue + store ------------------------------------------------
#pragma unroll
    for (int mt = 0; mt < 2; ++mt) {
#pragma unroll
        for (int nt = 0; nt < 4; ++nt) {
            const int col = n0 + wn * 64 + nt * 16 + l16;
            float bv = 0.f;
            if (EPI == EPI_BIAS || EPI == EPI_GELU_BIAS || EPI == EPI_RELU_BIAS)
                bv = bias[col];
#pragma unroll
            for (int r = 0; r < 8; ++r) {
                const int row = m0 + wm * 32 + mt * 16 + half * 8 + r;
                float v = acc[mt][nt][r];
                if (EPI == EPI_BIAS) {
                    v += bv;
                } else if (EPI == EPI_SILU) {
                    v = v / (1.0f + __expf(-v));
                } else if (EPI == EPI_GELU_BIAS) {
                    float t = v + bv;
                    v = 0.5f * t * (1.0f + erff(t * 0.70710678118654752f));
                } else if (EPI == EPI_RELU_BIAS) {
                    v = fmaxf(v + bv, 0.0f);
                }
                C[(size_t)row * N + col] = f2bf(v);
            }
        }
    }
}

// ---------------- elementwise gate: g = a * b (both bf16) --------------------
__global__ __launch_bounds__(256) void mul_kernel(
    const unsigned short* __restrict__ a, const unsigned short* __restrict__ b,
    unsigned short* __restrict__ out, size_t n)
{
    size_t i = (size_t)blockIdx.x * blockDim.x + threadIdx.x;
    size_t stride = (size_t)gridDim.x * blockDim.x;
    for (; i < n; i += stride) out[i] = f2bf(bf2f(a[i]) * bf2f(b[i]));
}

// ---------------- out = LayerNorm(a + b) * g + beta --------------------------
__global__ __launch_bounds__(256) void add_ln_kernel(
    const unsigned short* __restrict__ a, const unsigned short* __restrict__ b,
    const float* __restrict__ g, const float* __restrict__ beta,
    unsigned short* __restrict__ out)
{
    const int row  = blockIdx.x;
    const int tid  = threadIdx.x;
    const int lane = tid & 31;
    const int wave = tid >> 5;
    __shared__ float s_sum[8], s_sq[8];

    float v[4];
    float sum = 0.f, sq = 0.f;
#pragma unroll
    for (int i = 0; i < 4; ++i) {
        int h = tid + i * 256;
        size_t idx = (size_t)row * HDIM + h;
        v[i] = bf2f(a[idx]) + bf2f(b[idx]);
        sum += v[i];
        sq  += v[i] * v[i];
    }
    for (int off = 16; off > 0; off >>= 1) {
        sum += __shfl_xor(sum, off, 32);
        sq  += __shfl_xor(sq,  off, 32);
    }
    if (lane == 0) { s_sum[wave] = sum; s_sq[wave] = sq; }
    __syncthreads();
    float ts = 0.f, tq = 0.f;
#pragma unroll
    for (int w = 0; w < 8; ++w) { ts += s_sum[w]; tq += s_sq[w]; }
    const float mean = ts * (1.0f / HDIM);
    const float var  = tq * (1.0f / HDIM) - mean * mean;
    const float rstd = rsqrtf(var + 1e-5f);
#pragma unroll
    for (int i = 0; i < 4; ++i) {
        int h = tid + i * 256;
        size_t idx = (size_t)row * HDIM + h;
        out[idx] = f2bf((v[i] - mean) * rstd * g[h] + beta[h]);
    }
}

// ---------------- top-2 combine into fp32 output -----------------------------
__global__ __launch_bounds__(256) void combine_kernel(
    const unsigned short* __restrict__ e0, const unsigned short* __restrict__ e1,
    const unsigned short* __restrict__ e2, const unsigned short* __restrict__ e3,
    const float* __restrict__ w, const int* __restrict__ sel,
    float* __restrict__ out, size_t total)
{
    size_t i = (size_t)blockIdx.x * blockDim.x + threadIdx.x;
    size_t stride = (size_t)gridDim.x * blockDim.x;
    for (; i < total; i += stride) {
        size_t token = i / HDIM;
        int s0 = sel[token * 2 + 0];
        int s1 = sel[token * 2 + 1];
        const unsigned short* p0 = (s0 == 0) ? e0 : (s0 == 1) ? e1 : (s0 == 2) ? e2 : e3;
        const unsigned short* p1 = (s1 == 0) ? e0 : (s1 == 1) ? e1 : (s1 == 2) ? e2 : e3;
        out[i] = w[token * 2 + 0] * bf2f(p0[i]) + w[token * 2 + 1] * bf2f(p1[i]);
    }
}

// ---------------------------------------------------------------------------
// host side
// ---------------------------------------------------------------------------
static inline void* carve(char** p, size_t bytes) {
    void* r = (void*)*p;
    *p += (bytes + 255) & ~(size_t)255;
    return r;
}

static void launch_gemm(int epi, const unsigned short* A, const unsigned short* Bt,
                        const float* bias, unsigned short* C,
                        int M, int N, int K, hipStream_t s)
{
    dim3 grid(N / 128, M / 128), block(256);
    switch (epi) {
    case EPI_NONE:      gemm_bf16_kernel<EPI_NONE>     <<<grid, block, 0, s>>>(A, Bt, bias, C, M, N, K); break;
    case EPI_BIAS:      gemm_bf16_kernel<EPI_BIAS>     <<<grid, block, 0, s>>>(A, Bt, bias, C, M, N, K); break;
    case EPI_SILU:      gemm_bf16_kernel<EPI_SILU>     <<<grid, block, 0, s>>>(A, Bt, bias, C, M, N, K); break;
    case EPI_GELU_BIAS: gemm_bf16_kernel<EPI_GELU_BIAS><<<grid, block, 0, s>>>(A, Bt, bias, C, M, N, K); break;
    case EPI_RELU_BIAS: gemm_bf16_kernel<EPI_RELU_BIAS><<<grid, block, 0, s>>>(A, Bt, bias, C, M, N, K); break;
    }
}

static void launch_convert(const float* src, unsigned short* dst, size_t n, hipStream_t s)
{
    int blocks = (int)((n + 1023) / 1024);
    convert_bf16_kernel<<<blocks, 256, 0, s>>>(src, dst, n);
}

// weight [K,N] fp32 -> [N,K] bf16
static void launch_convert_tr(const float* src, unsigned short* dst, int K, int N, hipStream_t s)
{
    dim3 grid(K / 32, N / 32), block(256);
    convert_transpose_kernel<<<grid, block, 0, s>>>(src, dst, N, K);
}

extern "C" void kernel_launch(void* const* d_in, const int* in_sizes, int n_in,
                              void* d_out, int out_size, void* d_ws, size_t ws_size,
                              hipStream_t stream)
{
    (void)in_sizes; (void)n_in; (void)out_size; (void)ws_size;
    const int NT = NTOK, H = HDIM, I = IDIM;

    const float* x        = (const float*)d_in[0];
    const float* router_w = (const float*)d_in[1];
    const float* router_b = (const float*)d_in[2];
    const float* load_bal = (const float*)d_in[3];
    const float* sw_w1    = (const float*)d_in[4];
    const float* sw_w3    = (const float*)d_in[5];
    const float* sw_w2    = (const float*)d_in[6];
    const float* me_eq_w  = (const float*)d_in[7];
    const float* me_eq_b  = (const float*)d_in[8];
    const float* me_wv    = (const float*)d_in[9];
    const float* me_bv    = (const float*)d_in[10];
    const float* me_wo    = (const float*)d_in[11];
    const float* me_bo    = (const float*)d_in[12];
    const float* me_c1w   = (const float*)d_in[13];
    const float* me_c1b   = (const float*)d_in[14];
    const float* me_c2w   = (const float*)d_in[15];
    const float* me_c2b   = (const float*)d_in[16];
    const float* ce_syn_w = (const float*)d_in[17];
    const float* ce_syn_b = (const float*)d_in[18];
    const float* ce_wv    = (const float*)d_in[19];
    const float* ce_bv    = (const float*)d_in[20];
    const float* ce_wo    = (const float*)d_in[21];
    const float* ce_bo    = (const float*)d_in[22];
    const float* ce_n1g   = (const float*)d_in[23];
    const float* ce_n1b   = (const float*)d_in[24];
    const float* ce_f1w   = (const float*)d_in[25];
    const float* ce_f1b   = (const float*)d_in[26];
    const float* ce_f2w   = (const float*)d_in[27];
    const float* ce_f2b   = (const float*)d_in[28];
    const float* ce_n2g   = (const float*)d_in[29];
    const float* ce_n2b   = (const float*)d_in[30];
    const float* ce_gen_w = (const float*)d_in[31];
    const float* ce_gen_b = (const float*)d_in[32];
    const float* ml_w1    = (const float*)d_in[33];
    const float* ml_b1    = (const float*)d_in[34];
    const float* ml_w2    = (const float*)d_in[35];
    const float* ml_b2    = (const float*)d_in[36];

    // ---- workspace layout (bf16 stored as ushort; weights transposed) ----
    char* wp = (char*)d_ws;
    typedef unsigned short us;
    us* XB   = (us*)carve(&wp, (size_t)NT * H * 2);
    us* Wsw1 = (us*)carve(&wp, (size_t)H * I * 2);
    us* Wsw3 = (us*)carve(&wp, (size_t)H * I * 2);
    us* Wsw2 = (us*)carve(&wp, (size_t)I * H * 2);
    us* Weq  = (us*)carve(&wp, (size_t)H * H * 2);
    us* Wwv  = (us*)carve(&wp, (size_t)H * H * 2);
    us* Wwo  = (us*)carve(&wp, (size_t)H * H * 2);
    us* Wc1  = (us*)carve(&wp, (size_t)H * 2 * H * 2);
    us* Wc2  = (us*)carve(&wp, (size_t)2 * H * H * 2);
    us* Wsyn = (us*)carve(&wp, (size_t)H * H * 2);
    us* Wcwv = (us*)carve(&wp, (size_t)H * H * 2);
    us* Wcwo = (us*)carve(&wp, (size_t)H * H * 2);
    us* Wf1  = (us*)carve(&wp, (size_t)H * 2 * H * 2);
    us* Wf2  = (us*)carve(&wp, (size_t)2 * H * H * 2);
    us* Wgen = (us*)carve(&wp, (size_t)H * H * 2);
    us* Wml1 = (us*)carve(&wp, (size_t)H * I * 2);
    us* Wml2 = (us*)carve(&wp, (size_t)I * H * 2);
    us* TA   = (us*)carve(&wp, (size_t)NT * I * 2);
    us* TB   = (us*)carve(&wp, (size_t)NT * I * 2);
    us* H1   = (us*)carve(&wp, (size_t)NT * H * 2);
    us* H2   = (us*)carve(&wp, (size_t)NT * H * 2);
    us* H3   = (us*)carve(&wp, (size_t)NT * H * 2);
    us* E0   = (us*)carve(&wp, (size_t)NT * H * 2);
    us* E1   = (us*)carve(&wp, (size_t)NT * H * 2);
    us* E2   = (us*)carve(&wp, (size_t)NT * H * 2);
    us* E3   = (us*)carve(&wp, (size_t)NT * H * 2);
    float* RW  = (float*)carve(&wp, (size_t)NT * 2 * 4);
    int*   SEL = (int*)  carve(&wp, (size_t)NT * 2 * 4);

    // ---- convert activations (linear) + weights (transposed) to bf16 ----
    launch_convert(x, XB, (size_t)NT * H, stream);
    launch_convert_tr(sw_w1,    Wsw1, H, I,     stream);   // [H,I] -> [I,H]
    launch_convert_tr(sw_w3,    Wsw3, H, I,     stream);
    launch_convert_tr(sw_w2,    Wsw2, I, H,     stream);   // [I,H] -> [H,I]
    launch_convert_tr(me_eq_w,  Weq,  H, H,     stream);
    launch_convert_tr(me_wv,    Wwv,  H, H,     stream);
    launch_convert_tr(me_wo,    Wwo,  H, H,     stream);
    launch_convert_tr(me_c1w,   Wc1,  H, 2 * H, stream);
    launch_convert_tr(me_c2w,   Wc2,  2 * H, H, stream);
    launch_convert_tr(ce_syn_w, Wsyn, H, H,     stream);
    launch_convert_tr(ce_wv,    Wcwv, H, H,     stream);
    launch_convert_tr(ce_wo,    Wcwo, H, H,     stream);
    launch_convert_tr(ce_f1w,   Wf1,  H, 2 * H, stream);
    launch_convert_tr(ce_f2w,   Wf2,  2 * H, H, stream);
    launch_convert_tr(ce_gen_w, Wgen, H, H,     stream);
    launch_convert_tr(ml_w1,    Wml1, H, I,     stream);
    launch_convert_tr(ml_w2,    Wml2, I, H,     stream);

    // ---- router (fp32) ----
    router_kernel<<<NT / 8, 256, 0, stream>>>(x, router_w, router_b, load_bal, RW, SEL);

    // ---- expert 0: SwiGLU ----
    launch_gemm(EPI_SILU, XB, Wsw1, nullptr, TA, NT, I, H, stream);
    launch_gemm(EPI_NONE, XB, Wsw3, nullptr, TB, NT, I, H, stream);
    {
        size_t n = (size_t)NT * I;
        mul_kernel<<<(int)((n + 1023) / 1024), 256, 0, stream>>>(TA, TB, TA, n);
    }
    launch_gemm(EPI_NONE, TA, Wsw2, nullptr, E0, NT, H, I, stream);

    // ---- expert 1: MathExpert ----
    launch_gemm(EPI_BIAS, XB, Weq, me_eq_b, H1, NT, H, H, stream);
    launch_gemm(EPI_BIAS, H1, Wwv, me_bv, H2, NT, H, H, stream);
    launch_gemm(EPI_BIAS, H2, Wwo, me_bo, H1, NT, H, H, stream);
    launch_gemm(EPI_GELU_BIAS, H1, Wc1, me_c1b, TA, NT, 2 * H, H, stream);
    launch_gemm(EPI_BIAS, TA, Wc2, me_c2b, E1, NT, H, 2 * H, stream);

    // ---- expert 2: CodeExpert ----
    launch_gemm(EPI_BIAS, XB, Wsyn, ce_syn_b, H1, NT, H, H, stream);
    launch_gemm(EPI_BIAS, H1, Wcwv, ce_bv, H2, NT, H, H, stream);
    launch_gemm(EPI_BIAS, H2, Wcwo, ce_bo, H3, NT, H, H, stream);
    add_ln_kernel<<<NT, 256, 0, stream>>>(H1, H3, ce_n1g, ce_n1b, H2);
    launch_gemm(EPI_RELU_BIAS, H2, Wf1, ce_f1b, TA, NT, 2 * H, H, stream);
    launch_gemm(EPI_BIAS, TA, Wf2, ce_f2b, H1, NT, H, 2 * H, stream);
    add_ln_kernel<<<NT, 256, 0, stream>>>(H2, H1, ce_n2g, ce_n2b, H3);
    launch_gemm(EPI_BIAS, H3, Wgen, ce_gen_b, E2, NT, H, H, stream);

    // ---- expert 3: GELU MLP ----
    launch_gemm(EPI_GELU_BIAS, XB, Wml1, ml_b1, TB, NT, I, H, stream);
    launch_gemm(EPI_BIAS, TB, Wml2, ml_b2, E3, NT, H, I, stream);

    // ---- top-2 combine ----
    {
        size_t n = (size_t)NT * H;
        combine_kernel<<<(int)((n + 1023) / 1024), 256, 0, stream>>>(
            E0, E1, E2, E3, RW, SEL, (float*)d_out, n);
    }
}
